// TDNNKalmanNet_19344532701426
// MI455X (gfx1250) — matrix operand order repaired and checked by reading, not compiled, not run on hardware
//
#include <hip/hip_runtime.h>
#include <math.h>

// ---------------- problem constants ----------------
#define Bz 512
#define Qz 256
#define Dz 256
#define Hz 256
#define WOUT 32
#define PI_F 3.14159265358979f

typedef __attribute__((ext_vector_type(16))) __bf16 bf16x16;
typedef __attribute__((ext_vector_type(8)))  __bf16 bf16x8;
typedef __attribute__((ext_vector_type(8)))  float  f32x8;

// ---------------- workspace layout (bytes) ----------------
#define OFF_FEATS 0ull                                   // bf16 [B,Q,2D]  128MiB (reused as Kbuf)
#define OFF_H1    (OFF_FEATS + (size_t)Bz*Qz*2*Dz*2)     // bf16 [B,Q,H]
#define OFF_HSEQ  (OFF_H1    + (size_t)Bz*Qz*Hz*2)       // bf16 [B,Q,H]
#define OFF_RHO   (OFF_HSEQ  + (size_t)Bz*Qz*Hz*2)       // f32  [B,Q]
#define OFF_PHI   (OFF_RHO   + (size_t)Bz*Qz*4)          // f32  [B,Q]
#define OFF_CURR  (OFF_PHI   + (size_t)Bz*Qz*4)          // f32  [B,D]
#define OFF_HR    (OFF_CURR  + (size_t)Bz*Dz*4)          // f32  [B,H]
#define OFF_W1    (OFF_HR    + (size_t)Bz*Hz*4)          // bf16 [H,1536]
#define OFF_W2    (OFF_W1    + (size_t)Hz*1536*2)        // bf16 [H,768]
#define OFF_WG    (OFF_W2    + (size_t)Hz*768*2)         // bf16 [D,H]
#define OFF_WROLL (OFF_WG    + (size_t)Dz*Hz*2)          // bf16 [H,512]
#define OFF_KBUF  OFF_FEATS                              // bf16 [B,Q,D] (reuses feats)

__device__ __forceinline__ float geluf(float x) {
    return 0.5f * x * (1.0f + erff(x * 0.70710678118654752f));
}
__device__ __forceinline__ float sigmoidf_(float x) {
    return 1.0f / (1.0f + __expf(-x));
}
__device__ __forceinline__ float wave_red(float v) {
    #pragma unroll
    for (int m = 16; m >= 1; m >>= 1) v += __shfl_xor(v, m, 32);
    return v;
}

// ---- CDNA5 async global->LDS copy (ASYNCcnt-tracked), inline asm per ISA 08 ----
__device__ __forceinline__ void async_ld_b64(unsigned lds_addr, const void* gptr) {
    asm volatile("global_load_async_to_lds_b64 %0, %1, off"
                 :: "v"(lds_addr), "v"((unsigned long long)(uintptr_t)gptr)
                 : "memory");
}
__device__ __forceinline__ void wait_async_le1() {
    asm volatile("s_wait_asynccnt 0x1" ::: "memory");
}
__device__ __forceinline__ void wait_async_le0() {
    asm volatile("s_wait_asynccnt 0x0" ::: "memory");
}

// ---------------- weight repack to bf16, K-contiguous per output column ----------------
__global__ __launch_bounds__(256) void prep_weights(
    const float* __restrict__ c1, const float* __restrict__ c2,
    const float* __restrict__ wg, const float* __restrict__ wroll,
    __bf16* __restrict__ w1, __bf16* __restrict__ w2,
    __bf16* __restrict__ wgb, __bf16* __restrict__ wrb)
{
    int tid = blockIdx.x * 256 + threadIdx.x;
    if (tid < 393216) {                       // conv1: [H,1536], kk = tap*512 + c
        int h = tid / 1536, kk = tid % 1536;
        int tap = kk >> 9, c = kk & 511;
        w1[(size_t)h * 1536 + kk] = (__bf16)c1[(size_t)h * 1536 + c * 3 + tap];
    } else if (tid < 393216 + 196608) {       // conv2: [H,768], kk = tap*256 + c
        int i = tid - 393216;
        int h = i / 768, kk = i % 768;
        int tap = kk >> 8, c = kk & 255;
        w2[(size_t)h * 768 + kk] = (__bf16)c2[(size_t)h * 768 + c * 3 + tap];
    } else if (tid < 393216 + 196608 + 65536) { // Wg: [D,H] row-major already K-contig
        int i = tid - (393216 + 196608);
        wgb[i] = (__bf16)wg[i];
    } else if (tid < 393216 + 196608 + 65536 + 131072) { // Wroll: [H,512]
        int i = tid - (393216 + 196608 + 65536);
        wrb[i] = (__bf16)wroll[i];
    }
}

// ---------------- feats = concat(x, dy) in bf16 ----------------
__global__ __launch_bounds__(256) void build_feats(const float* __restrict__ x,
                                                   __bf16* __restrict__ feats)
{
    size_t tid = (size_t)blockIdx.x * 256 + threadIdx.x; // B*Q*D
    int d = (int)(tid & 255);
    size_t bq = tid >> 8;
    int q = (int)(bq & 255);
    float xv = x[tid];
    float dy = (q == 0) ? 0.0f : xv - x[tid - 256];
    size_t o = bq * 512;
    feats[o + d]        = (__bf16)xv;
    feats[o + 256 + d]  = (__bf16)dy;
}

// ---------------- implicit-GEMM conv / GEMM with WMMA bf16 ----------------
// Block = 512 threads (16 waves) computes a 64(M) x 256(N) output tile.
// A tile (64x32 bf16 per k-chunk) is staged in LDS via async global->LDS loads,
// double-buffered; wave w owns N-cols [w*16,w*16+16) and all 4 M-subtiles, so each
// B fragment is register-reused 4x. Edge-clamped q offsets per tap give conv padding.
// act: 0 = GELU, 1 = sigmoid. Output bf16 [M,256].
__global__ __launch_bounds__(512) void wmma_conv_gemm(
    const __bf16* __restrict__ src, int Cshift, int dil, int Kdim,
    const __bf16* __restrict__ wbuf, const float* __restrict__ bias,
    __bf16* __restrict__ dst, int act)
{
    __shared__ alignas(16) __bf16 sA[2][64 * 32];   // 2 x 4KB
    const int C = 1 << Cshift;
    const int tid = threadIdx.x;
    const int wave = tid >> 5, lane = tid & 31;
    const int khalf = lane >> 4, l15 = lane & 15;
    const int tileM = blockIdx.x;                   // 64-row tile

    // staging role: thread -> (row, 8-byte segment) of the A chunk
    const int srow = tid >> 3;                      // 0..63
    const int sseg = tid & 7;                       // 0..7 (8B each)
    const int sm = tileM * 64 + srow;
    const int sb = sm >> 8, sq = sm & 255;
    const unsigned ldsbase = (unsigned)(uintptr_t)(&sA[0][0]);
    const int nchunks = Kdim >> 5;

    const int col = wave * 16 + l15;
    const __bf16* wrow = wbuf + (size_t)col * Kdim + khalf * 16;
    f32x8 acc[4] = {};

    // stage chunk 0 (tap 0 -> q offset -dil, clamped)
    {
        int qq = sq - dil; qq = qq < 0 ? 0 : (qq > 255 ? 255 : qq);
        const __bf16* gp = src + (((size_t)(sb << 8) + qq) << Cshift) + sseg * 4;
        async_ld_b64(ldsbase + srow * 64 + sseg * 8, gp);
    }
    for (int i = 0; i < nchunks; i++) {
        if (i + 1 < nchunks) {
            int kbase = (i + 1) << 5;
            int tap = kbase >> Cshift;              // chunks never straddle taps
            int cbase = kbase & (C - 1);
            int qq = sq + (tap - 1) * dil;
            qq = qq < 0 ? 0 : (qq > 255 ? 255 : qq);
            const __bf16* gp = src + (((size_t)(sb << 8) + qq) << Cshift) + cbase + sseg * 4;
            async_ld_b64(ldsbase + ((unsigned)((i + 1) & 1)) * 4096u + srow * 64 + sseg * 8, gp);
            wait_async_le1();                       // oldest (chunk i) has landed
        } else {
            wait_async_le0();
        }
        __syncthreads();                            // publish chunk i to all waves

        const __bf16* sAc = &sA[i & 1][0];
        bf16x16 bfrag = *(const bf16x16*)(wrow + (i << 5));
        #pragma unroll
        for (int s = 0; s < 4; s++) {
            const __bf16* ap = sAc + (s * 16 + l15) * 32 + khalf * 16;
            bf16x8 lo = *(const bf16x8*)(ap);
            bf16x8 hi = *(const bf16x8*)(ap + 8);
            bf16x16 afrag = __builtin_shufflevector(lo, hi,
                0,1,2,3,4,5,6,7,8,9,10,11,12,13,14,15);
            acc[s] = __builtin_amdgcn_wmma_f32_16x16x32_bf16(
                false, afrag, false, bfrag, (short)0, acc[s], false, false);
        }
        __syncthreads();                            // done reading before next overwrite
    }

    float bv = bias[col];
    #pragma unroll
    for (int s = 0; s < 4; s++) {
        #pragma unroll
        for (int r = 0; r < 8; r++) {
            int row = tileM * 64 + s * 16 + r + khalf * 8;
            float v = acc[s][r] + bv;
            float o = (act == 0) ? geluf(v) : sigmoidf_(v);
            dst[(size_t)row * 256 + col] = (__bf16)o;
        }
    }
}

// ---------------- LayerNorm over H (in place, bf16) + rp head -> rho/phi ----------------
__global__ __launch_bounds__(256) void ln_rp(
    __bf16* __restrict__ hseq, const float* __restrict__ g, const float* __restrict__ b_,
    const float* __restrict__ Wrp, const float* __restrict__ brp,
    float* __restrict__ rho, float* __restrict__ phi)
{
    int m = blockIdx.x;                 // B*Q rows
    int h = threadIdx.x, lane = h & 31, wid = h >> 5;
    __shared__ float rA[8], rB[8];
    float v = (float)hseq[(size_t)m * 256 + h];
    float s = wave_red(v), s2 = wave_red(v * v);
    if (lane == 0) { rA[wid] = s; rB[wid] = s2; }
    __syncthreads();
    float tot = 0.f, tot2 = 0.f;
    #pragma unroll
    for (int i = 0; i < 8; i++) { tot += rA[i]; tot2 += rB[i]; }
    float mean = tot * (1.0f / 256.0f);
    float var = tot2 * (1.0f / 256.0f) - mean * mean;
    float hn = (v - mean) * rsqrtf(var + 1e-5f) * g[h] + b_[h];
    hseq[(size_t)m * 256 + h] = (__bf16)hn;
    float d0 = wave_red(hn * Wrp[h]);
    float d1 = wave_red(hn * Wrp[256 + h]);
    __syncthreads();
    if (lane == 0) { rA[wid] = d0; rB[wid] = d1; }
    __syncthreads();
    if (h == 0) {
        float t0 = 0.f, t1 = 0.f;
        #pragma unroll
        for (int i = 0; i < 8; i++) { t0 += rA[i]; t1 += rB[i]; }
        rho[m] = 1.2f * sigmoidf_(t0 + brp[0]);
        phi[m] = PI_F * tanhf(t1 + brp[1]);
    }
}

// ---------------- sequential Kalman scan: one thread per (batch, rotation pair) ----------------
__global__ __launch_bounds__(256) void kalman_scan(
    const float* __restrict__ x_in, const __bf16* __restrict__ kbuf,
    const float* __restrict__ rho, const float* __restrict__ phi,
    float* __restrict__ curr)
{
    int tid = blockIdx.x * 256 + threadIdx.x;   // B*128
    int b = tid >> 7, mm = tid & 127;
    float xr = 0.f, xi = 0.f;
    for (int q = 0; q < 256; q++) {
        int bq = (b << 8) + q;
        float r = rho[bq];
        float sn, cs; __sincosf(phi[bq], &sn, &cs);
        float prr = r * (cs * xr - sn * xi);
        float pri = r * (sn * xr + cs * xi);
        size_t o = (size_t)bq * 256;
        float yre = x_in[o + mm],        yim = x_in[o + 128 + mm];
        float kre = (float)kbuf[o + mm], kim = (float)kbuf[o + 128 + mm];
        xr = prr + kre * (yre - prr);
        xi = pri + kim * (yim - pri);
    }
    curr[(size_t)b * 256 + mm]       = xr;
    curr[(size_t)b * 256 + 128 + mm] = xi;
}

// ---------------- rollout init: h_r = h_seq[:, Q-1, :] ----------------
__global__ __launch_bounds__(256) void hr_init(const __bf16* __restrict__ hseq,
                                               float* __restrict__ h_r)
{
    int tid = blockIdx.x * 256 + threadIdx.x;   // B*H
    int b = tid >> 8, h = tid & 255;
    h_r[tid] = (float)hseq[(((size_t)(b << 8)) + 255) * 256 + h];
}

// ---------------- one rollout step: WMMA GEMM + GELU + LN + rp + rotate + emit ----------------
__global__ __launch_bounds__(512) void rollout_step(
    const __bf16* __restrict__ wroll, const float* __restrict__ broll,
    const float* __restrict__ g, const float* __restrict__ beta,
    const float* __restrict__ Wrpr, const float* __restrict__ brpr,
    float* __restrict__ h_r, float* __restrict__ curr,
    float* __restrict__ out, int t)
{
    __shared__ float zbuf[16 * 256];
    __shared__ float srho[16], sphi[16];
    int wave = threadIdx.x >> 5, lane = threadIdx.x & 31;
    int khalf = lane >> 4, l15 = lane & 15;
    int row0 = blockIdx.x * 16;                 // 16 batches per block

    // ---- Z = [h_r, curr] @ Wroll^T : wave `wave` computes cols [wave*16, wave*16+16)
    int mrow = row0 + l15;
    int col = wave * 16 + l15;
    const float* hrow = h_r + (size_t)mrow * 256;
    const float* crow = curr + (size_t)mrow * 256;
    f32x8 acc = {};
    #pragma unroll
    for (int kb = 0; kb < 512; kb += 32) {
        // each 32-chunk lies entirely in h_r (kb<256) or curr (kb>=256)
        const float* srcp = (kb < 256) ? (hrow + kb) : (crow + (kb - 256));
        float4 a0 = *(const float4*)(srcp + khalf * 16);
        float4 a1 = *(const float4*)(srcp + khalf * 16 + 4);
        float4 a2 = *(const float4*)(srcp + khalf * 16 + 8);
        float4 a3 = *(const float4*)(srcp + khalf * 16 + 12);
        bf16x16 afrag;
        afrag[0]=(__bf16)a0.x; afrag[1]=(__bf16)a0.y; afrag[2]=(__bf16)a0.z; afrag[3]=(__bf16)a0.w;
        afrag[4]=(__bf16)a1.x; afrag[5]=(__bf16)a1.y; afrag[6]=(__bf16)a1.z; afrag[7]=(__bf16)a1.w;
        afrag[8]=(__bf16)a2.x; afrag[9]=(__bf16)a2.y; afrag[10]=(__bf16)a2.z; afrag[11]=(__bf16)a2.w;
        afrag[12]=(__bf16)a3.x; afrag[13]=(__bf16)a3.y; afrag[14]=(__bf16)a3.z; afrag[15]=(__bf16)a3.w;
        bf16x16 bfrag = *(const bf16x16*)(wroll + (size_t)col * 512 + kb + khalf * 16);
        acc = __builtin_amdgcn_wmma_f32_16x16x32_bf16(
            false, afrag, false, bfrag, (short)0, acc, false, false);
    }
    float bv = broll[col];
    #pragma unroll
    for (int r = 0; r < 8; r++) {
        int rr = r + khalf * 8;
        zbuf[rr * 256 + col] = geluf(acc[r] + bv);
    }
    __syncthreads();

    // ---- per-row LN (wave w owns row w), fused Wrpr projection
    {
        int row = wave;
        float vals[8]; float s = 0.f, s2 = 0.f;
        #pragma unroll
        for (int j = 0; j < 8; j++) {
            float v = zbuf[row * 256 + lane + j * 32];
            vals[j] = v; s += v; s2 += v * v;
        }
        s = wave_red(s); s2 = wave_red(s2);
        float mean = s * (1.0f / 256.0f);
        float var = s2 * (1.0f / 256.0f) - mean * mean;
        float inv = rsqrtf(var + 1e-5f);
        int gb = row0 + row;
        float d0 = 0.f, d1 = 0.f;
        #pragma unroll
        for (int j = 0; j < 8; j++) {
            int h = lane + j * 32;
            float hn = (vals[j] - mean) * inv * g[h] + beta[h];
            h_r[(size_t)gb * 256 + h] = hn;
            d0 += hn * Wrpr[h];
            d1 += hn * Wrpr[256 + h];
        }
        d0 = wave_red(d0); d1 = wave_red(d1);
        if (lane == 0) {
            srho[row] = 1.2f * sigmoidf_(d0 + brpr[0]);
            sphi[row] = PI_F * tanhf(d1 + brpr[1]);
        }
    }
    __syncthreads();

    // ---- rotate curr and emit prediction for step t
    #pragma unroll
    for (int it = 0; it < 4; it++) {
        int idx = threadIdx.x + it * 512;       // 16 rows * 128 pairs
        int row = idx >> 7, mm = idx & 127;
        int gb = row0 + row;
        float re = curr[(size_t)gb * 256 + mm];
        float im = curr[(size_t)gb * 256 + 128 + mm];
        float sn, cs; __sincosf(sphi[row], &sn, &cs);
        float rho = srho[row];
        float nre = rho * (cs * re - sn * im);
        float nim = rho * (sn * re + cs * im);
        curr[(size_t)gb * 256 + mm]       = nre;
        curr[(size_t)gb * 256 + 128 + mm] = nim;
        out[((size_t)gb * WOUT + t) * 256 + mm]       = nre;
        out[((size_t)gb * WOUT + t) * 256 + 128 + mm] = nim;
    }
}

// ---------------- host launcher ----------------
extern "C" void kernel_launch(void* const* d_in, const int* in_sizes, int n_in,
                              void* d_out, int out_size, void* d_ws, size_t ws_size,
                              hipStream_t stream)
{
    const float* x_in    = (const float*)d_in[0];
    const float* conv1_w = (const float*)d_in[1];
    const float* conv1_b = (const float*)d_in[2];
    const float* conv2_w = (const float*)d_in[3];
    const float* conv2_b = (const float*)d_in[4];
    const float* enc_g   = (const float*)d_in[5];
    const float* enc_b   = (const float*)d_in[6];
    const float* Wg      = (const float*)d_in[7];
    const float* bg      = (const float*)d_in[8];
    const float* Wrp     = (const float*)d_in[9];
    const float* brp     = (const float*)d_in[10];
    const float* Wroll   = (const float*)d_in[11];
    const float* broll   = (const float*)d_in[12];
    const float* rg      = (const float*)d_in[13];
    const float* rb      = (const float*)d_in[14];
    const float* Wrpr    = (const float*)d_in[15];
    const float* brpr    = (const float*)d_in[16];
    (void)in_sizes; (void)n_in; (void)out_size; (void)ws_size;

    char* ws = (char*)d_ws;
    __bf16* feats = (__bf16*)(ws + OFF_FEATS);
    __bf16* h1    = (__bf16*)(ws + OFF_H1);
    __bf16* hseq  = (__bf16*)(ws + OFF_HSEQ);
    float*  rho   = (float*)(ws + OFF_RHO);
    float*  phi   = (float*)(ws + OFF_PHI);
    float*  curr  = (float*)(ws + OFF_CURR);
    float*  h_r   = (float*)(ws + OFF_HR);
    __bf16* w1    = (__bf16*)(ws + OFF_W1);
    __bf16* w2    = (__bf16*)(ws + OFF_W2);
    __bf16* wgb   = (__bf16*)(ws + OFF_WG);
    __bf16* wrb   = (__bf16*)(ws + OFF_WROLL);
    __bf16* kbuf  = (__bf16*)(ws + OFF_KBUF);
    float*  out   = (float*)d_out;

    // 1. repack weights to bf16
    prep_weights<<<3072, 256, 0, stream>>>(conv1_w, conv2_w, Wg, Wroll, w1, w2, wgb, wrb);
    // 2. feats = [x, dx]
    build_feats<<<(Bz * Qz * Dz) / 256, 256, 0, stream>>>(x_in, feats);
    // 3. conv1 (dil=1, C=512, K=1536) + GELU -> h1     (2048 blocks of 64 rows)
    wmma_conv_gemm<<<2048, 512, 0, stream>>>(feats, 9, 1, 1536, w1, conv1_b, h1, 0);
    // 4. conv2 (dil=2, C=256, K=768) + GELU -> hseq (pre-LN)
    wmma_conv_gemm<<<2048, 512, 0, stream>>>(h1, 8, 2, 768, w2, conv2_b, hseq, 0);
    // 5. LN in place + rho/phi head
    ln_rp<<<Bz * Qz, 256, 0, stream>>>(hseq, enc_g, enc_b, Wrp, brp, rho, phi);
    // 6. K_seq gate GEMM (C=256, K=256, dil=0) + sigmoid -> kbuf (reuses feats region)
    wmma_conv_gemm<<<2048, 512, 0, stream>>>(hseq, 8, 0, 256, wgb, bg, kbuf, 1);
    // 7. h_r init from last timestep
    hr_init<<<512, 256, 0, stream>>>(hseq, h_r);
    // 8. Kalman scan -> curr
    kalman_scan<<<256, 256, 0, stream>>>(x_in, kbuf, rho, phi, curr);
    // 9. 32 rollout steps
    for (int t = 0; t < WOUT; t++) {
        rollout_step<<<Bz / 16, 512, 0, stream>>>(wrb, broll, rg, rb, Wrpr, brpr,
                                                  h_r, curr, out, t);
    }
}
